// SConv_15290083574384
// MI455X (gfx1250) — matrix operand order repaired and checked by reading, not compiled, not run on hardware
//
#include <hip/hip_runtime.h>

#define BN_EPS 1e-5f

typedef __attribute__((ext_vector_type(16))) __bf16 v16bf;
typedef __attribute__((ext_vector_type(8)))  float  v8f;
typedef __attribute__((ext_vector_type(4)))  unsigned int v4u;

union Frag { v16bf bf; v4u q[2]; };

// ---- fp32 -> packed bf16x2, cheapest available path
#if __has_builtin(__builtin_amdgcn_cvt_pk_bf16_f32)
typedef __bf16 bf16x2 __attribute__((ext_vector_type(2)));
__device__ __forceinline__ unsigned int pack2bf(float lo, float hi) {
  union { bf16x2 v; unsigned int u; } cv;
  cv.v = __builtin_amdgcn_cvt_pk_bf16_f32(lo, hi);
  return cv.u;
}
#elif __has_builtin(__builtin_amdgcn_perm)
__device__ __forceinline__ unsigned int pack2bf(float lo, float hi) {
  // round-half-up on magnitude, then one v_perm_b32 assembles {hi16(b),hi16(a)}
  const unsigned int a = __float_as_uint(lo) + 0x8000u;
  const unsigned int b = __float_as_uint(hi) + 0x8000u;
  return __builtin_amdgcn_perm(b, a, 0x07060302u);
}
#else
__device__ __forceinline__ unsigned short f2bf(float f) {
  unsigned int u = __float_as_uint(f);
  u += 0x7FFFu + ((u >> 16) & 1u);   // round-to-nearest-even
  return (unsigned short)(u >> 16);
}
__device__ __forceinline__ unsigned int pack2bf(float lo, float hi) {
  return (unsigned int)f2bf(lo) | ((unsigned int)f2bf(hi) << 16);
}
#endif

__device__ __forceinline__ float silu_f(float v) {
  return v / (1.0f + __expf(-v));
}

// Problem dimensions (fixed by the reference)
#define BB 16
#define C1 128
#define C2 256
#define HH 128
#define WW 128

// LDS tiles: rows of 32 bf16 payload padded to 40 bf16 (80B, 16B-aligned,
// conflict-free b128 fragment reads). Double buffered.
//   A: 128 rows (M) x 32 (K)    B: 64 rows (N) x 32 (K)
#define ROWDW 20   // dwords per row (80B)

__device__ __forceinline__ v8f wmma_bf16(const Frag& a, const Frag& b, v8f c) {
  return __builtin_amdgcn_wmma_f32_16x16x32_bf16(
      false, a.bf, false, b.bf, (short)0, c, false, false);
}

// ============================================================
// Kernel A: y = silu(bn(conv3x3(x, w)))
// grid(B*H*(W/64)=4096, C2/128=2), block 256 (8 waves: 4 M x 2 N)
// Each wave: 32(M) x 32(N) output = 4 accumulators.
// K-loop: 36 steps of 32 channels at fixed (kh,kw).
// ============================================================
__global__ __launch_bounds__(256) void conv3_bn_silu(
    const float* __restrict__ x, const float* __restrict__ wconv,
    const float* __restrict__ gamma, const float* __restrict__ beta,
    const float* __restrict__ bmean, const float* __restrict__ bvar,
    float* __restrict__ y)
{
  __shared__ unsigned int As32[2][128 * ROWDW];
  __shared__ unsigned int Bs32[2][64 * ROWDW];

  const int sb   = blockIdx.x;
  const int wblk = sb & 1;
  const int h    = (sb >> 1) & (HH - 1);
  const int b    = sb >> 8;
  const int w0   = wblk * 64;
  const int m0   = blockIdx.y * 128;

  const int t     = threadIdx.x;
  const int lane  = t & 31;
  const int wave  = t >> 5;
  const int mwave = wave & 3;     // 32-row M slice
  const int nwave = wave >> 2;    // 32-col N slice
  const int half  = lane >> 4;
  const int lrow  = lane & 15;

  // staging thread mapping
  const int a_cp  = t & 15;       // c-pair (c = 2*a_cp)
  const int a_mt  = t >> 4;       // m = p*16 + a_mt
  const int b_n   = t & 63;       // pixel within tile
  const int b_cpb = t >> 6;       // c-pair = p*4 + b_cpb

  float aReg[8][2];
  float bReg[4][2];

  auto load_step = [&](int kh, int kw, int cb) {
    const int c0 = cb * 32;
    const int hsrc = h + kh - 1;
    const int hc   = min(max(hsrc, 0), HH - 1);
    const bool hok = (unsigned)hsrc < (unsigned)HH;
    // ---- B: x[b, c0+c, hsrc, w0+n+kw-1], branch-free zero pad
    const int wsrc = w0 + b_n + kw - 1;
    const int wc   = min(max(wsrc, 0), WW - 1);
    const bool ok  = hok && ((unsigned)wsrc < (unsigned)WW);
#pragma unroll
    for (int p = 0; p < 4; ++p) {
      const int c = (p * 4 + b_cpb) * 2;
      const float* px = x + ((b * C1 + c0 + c) * HH + hc) * WW + wc;
      const float v0 = px[0];
      const float v1 = px[HH * WW];
      bReg[p][0] = ok ? v0 : 0.0f;
      bReg[p][1] = ok ? v1 : 0.0f;
    }
    // ---- A: wconv[m0+m, c0+2*a_cp(+1), kh, kw]  (stride 9 between channels)
#pragma unroll
    for (int p = 0; p < 8; ++p) {
      const int m = p * 16 + a_mt;
      const float* pw =
          wconv + ((m0 + m) * C1 + c0 + a_cp * 2) * 9 + kh * 3 + kw;
      aReg[p][0] = pw[0];
      aReg[p][1] = pw[9];
    }
  };

  auto store_step = [&](int buf) {
#pragma unroll
    for (int p = 0; p < 4; ++p)
      Bs32[buf][b_n * ROWDW + p * 4 + b_cpb] = pack2bf(bReg[p][0], bReg[p][1]);
#pragma unroll
    for (int p = 0; p < 8; ++p)
      As32[buf][(p * 16 + a_mt) * ROWDW + a_cp] = pack2bf(aReg[p][0], aReg[p][1]);
  };

  v8f acc[2][2] = {{{}, {}}, {{}, {}}};

  auto mm_step = [&](int buf) {
    const unsigned int* A = As32[buf];
    const unsigned int* B = Bs32[buf];
    Frag fa0, fa1, fb0, fb1;
    const int rowA = mwave * 32 + lrow;
    const v4u* pa0 = (const v4u*)(A + rowA * ROWDW + half * 4);
    fa0.q[0] = pa0[0]; fa0.q[1] = pa0[2];                       // +32B
    const v4u* pa1 = (const v4u*)(A + (rowA + 16) * ROWDW + half * 4);
    fa1.q[0] = pa1[0]; fa1.q[1] = pa1[2];
    const int rowB = nwave * 32 + lrow;
    const v4u* pb0 = (const v4u*)(B + rowB * ROWDW + half * 4);
    fb0.q[0] = pb0[0]; fb0.q[1] = pb0[2];                       // +32B
    const v4u* pb1 = (const v4u*)(B + (rowB + 16) * ROWDW + half * 4);
    fb1.q[0] = pb1[0]; fb1.q[1] = pb1[2];
    acc[0][0] = wmma_bf16(fa0, fb0, acc[0][0]);
    acc[0][1] = wmma_bf16(fa0, fb1, acc[0][1]);
    acc[1][0] = wmma_bf16(fa1, fb0, acc[1][0]);
    acc[1][1] = wmma_bf16(fa1, fb1, acc[1][1]);
  };

  // ---- software pipeline over 36 K-steps, double-buffered LDS.
  // (kh,kw,cb) advanced incrementally: no div/mod in the loop.
  int nkh = 0, nkw = 0, ncb = 0;
  load_step(0, 0, 0);
  store_step(0);
  for (int s = 0; s < 36; ++s) {
    if (++ncb == 4) { ncb = 0; if (++nkw == 3) { nkw = 0; ++nkh; } }
    if (s + 1 < 36) load_step(nkh, nkw, ncb);   // global -> VGPR prefetch
    __syncthreads();                             // buf[s&1] stores visible
    mm_step(s & 1);
    if (s + 1 < 36) store_step((s + 1) & 1);     // fill other buffer
  }

  // ---- epilogue: BN + SiLU. f32 C/D layout: lanes0-15 M=r, lanes16-31 M=8+r
  const int col  = w0 + nwave * 32 + lrow;
#pragma unroll
  for (int mi = 0; mi < 2; ++mi) {
#pragma unroll
    for (int r = 0; r < 8; ++r) {
      const int ch = m0 + mwave * 32 + mi * 16 + half * 8 + r;
      const float inv  = gamma[ch] / sqrtf(bvar[ch] + BN_EPS);
      const float bias = beta[ch] - bmean[ch] * inv;
      const int base = ((b * C2 + ch) * HH + h) * WW;
      y[base + col]      = silu_f(acc[mi][0][r] * inv + bias);
      y[base + col + 16] = silu_f(acc[mi][1][r] * inv + bias);
    }
  }
}

// ============================================================
// Kernel B: out = silu(bn(conv1x1(shift(y)) + b1) + y)
// grid(4096, C2/128=2), block 256. K-loop: 8 steps of 32 channels,
// shift folded into source addressing per 64-channel group.
// ============================================================
__global__ __launch_bounds__(256) void shift_conv1_bn_res_silu(
    const float* __restrict__ y, const float* __restrict__ w1,
    const float* __restrict__ b1,
    const float* __restrict__ gamma, const float* __restrict__ beta,
    const float* __restrict__ bmean, const float* __restrict__ bvar,
    float* __restrict__ out)
{
  __shared__ unsigned int As32[2][128 * ROWDW];
  __shared__ unsigned int Bs32[2][64 * ROWDW];

  const int sb   = blockIdx.x;
  const int wblk = sb & 1;
  const int h    = (sb >> 1) & (HH - 1);
  const int b    = sb >> 8;
  const int w0   = wblk * 64;
  const int m0   = blockIdx.y * 128;

  const int t     = threadIdx.x;
  const int lane  = t & 31;
  const int wave  = t >> 5;
  const int mwave = wave & 3;
  const int nwave = wave >> 2;
  const int half  = lane >> 4;
  const int lrow  = lane & 15;

  const int a_cp  = t & 15;
  const int a_mt  = t >> 4;
  const int b_n   = t & 63;
  const int b_cpb = t >> 6;

  float aReg[8][2];
  float bReg[4][2];

  auto load_step = [&](int s) {
    const int g = s >> 1, cb = s & 1;
    const int c0 = g * 64 + cb * 32;
    const int dh = (g == 1) ? 1 : ((g == 3) ? -1 : 0);
    const int dw = (g == 0) ? -1 : ((g == 2) ? 1 : 0);
    const int hsrc = h + dh;
    const int hc   = min(max(hsrc, 0), HH - 1);
    const bool hok = (unsigned)hsrc < (unsigned)HH;
    const int wsrc = w0 + b_n + dw;
    const int wc   = min(max(wsrc, 0), WW - 1);
    const bool ok  = hok && ((unsigned)wsrc < (unsigned)WW);
#pragma unroll
    for (int p = 0; p < 4; ++p) {
      const int c = (p * 4 + b_cpb) * 2;
      const float* py = y + ((b * C2 + c0 + c) * HH + hc) * WW + wc;
      const float v0 = py[0];
      const float v1 = py[HH * WW];
      bReg[p][0] = ok ? v0 : 0.0f;
      bReg[p][1] = ok ? v1 : 0.0f;
    }
    // 1x1 weights: channel pair is contiguous -> b64 load
#pragma unroll
    for (int p = 0; p < 8; ++p) {
      const int m = p * 16 + a_mt;
      const float2 wp =
          *(const float2*)(w1 + (m0 + m) * C2 + c0 + a_cp * 2);
      aReg[p][0] = wp.x;
      aReg[p][1] = wp.y;
    }
  };

  auto store_step = [&](int buf) {
#pragma unroll
    for (int p = 0; p < 4; ++p)
      Bs32[buf][b_n * ROWDW + p * 4 + b_cpb] = pack2bf(bReg[p][0], bReg[p][1]);
#pragma unroll
    for (int p = 0; p < 8; ++p)
      As32[buf][(p * 16 + a_mt) * ROWDW + a_cp] = pack2bf(aReg[p][0], aReg[p][1]);
  };

  v8f acc[2][2] = {{{}, {}}, {{}, {}}};

  auto mm_step = [&](int buf) {
    const unsigned int* A = As32[buf];
    const unsigned int* B = Bs32[buf];
    Frag fa0, fa1, fb0, fb1;
    const int rowA = mwave * 32 + lrow;
    const v4u* pa0 = (const v4u*)(A + rowA * ROWDW + half * 4);
    fa0.q[0] = pa0[0]; fa0.q[1] = pa0[2];
    const v4u* pa1 = (const v4u*)(A + (rowA + 16) * ROWDW + half * 4);
    fa1.q[0] = pa1[0]; fa1.q[1] = pa1[2];
    const int rowB = nwave * 32 + lrow;
    const v4u* pb0 = (const v4u*)(B + rowB * ROWDW + half * 4);
    fb0.q[0] = pb0[0]; fb0.q[1] = pb0[2];
    const v4u* pb1 = (const v4u*)(B + (rowB + 16) * ROWDW + half * 4);
    fb1.q[0] = pb1[0]; fb1.q[1] = pb1[2];
    acc[0][0] = wmma_bf16(fa0, fb0, acc[0][0]);
    acc[0][1] = wmma_bf16(fa0, fb1, acc[0][1]);
    acc[1][0] = wmma_bf16(fa1, fb0, acc[1][0]);
    acc[1][1] = wmma_bf16(fa1, fb1, acc[1][1]);
  };

  load_step(0);
  store_step(0);
  for (int s = 0; s < 8; ++s) {
    if (s + 1 < 8) load_step(s + 1);
    __syncthreads();
    mm_step(s & 1);
    if (s + 1 < 8) store_step((s + 1) & 1);
  }

  // ---- epilogue: +bias, BN, +identity(y), SiLU
  const int col = w0 + nwave * 32 + lrow;
#pragma unroll
  for (int mi = 0; mi < 2; ++mi) {
#pragma unroll
    for (int r = 0; r < 8; ++r) {
      const int ch = m0 + mwave * 32 + mi * 16 + half * 8 + r;
      const float inv  = gamma[ch] / sqrtf(bvar[ch] + BN_EPS);
      const float bias = beta[ch] - bmean[ch] * inv;
      const float bb   = b1[ch];
      const int base = ((b * C2 + ch) * HH + h) * WW;
      const float id0 = y[base + col];
      const float id1 = y[base + col + 16];
      out[base + col]      = silu_f((acc[mi][0][r] + bb) * inv + bias + id0);
      out[base + col + 16] = silu_f((acc[mi][1][r] + bb) * inv + bias + id1);
    }
  }
}

extern "C" void kernel_launch(void* const* d_in, const int* in_sizes, int n_in,
                              void* d_out, int out_size, void* d_ws, size_t ws_size,
                              hipStream_t stream) {
  (void)in_sizes; (void)n_in; (void)out_size; (void)ws_size;
  const float* x      = (const float*)d_in[0];
  const float* wconv  = (const float*)d_in[1];
  const float* w1x1   = (const float*)d_in[2];
  const float* b1x1   = (const float*)d_in[3];
  const float* gamma  = (const float*)d_in[4];
  const float* beta   = (const float*)d_in[5];
  const float* bmean  = (const float*)d_in[6];
  const float* bvar   = (const float*)d_in[7];
  float* outp = (float*)d_out;
  float* yws  = (float*)d_ws;   // y: 16*256*128*128 floats = 256 MB

  dim3 grid(BB * HH * (WW / 64), C2 / 128);   // (4096, 2)
  dim3 block(256);

  conv3_bn_silu<<<grid, block, 0, stream>>>(x, wconv, gamma, beta, bmean,
                                            bvar, yws);
  shift_conv1_bn_res_silu<<<grid, block, 0, stream>>>(yws, w1x1, b1x1, gamma,
                                                      beta, bmean, bvar, outp);
}
